// MHAttentionFusionThird_10299331576164
// MI455X (gfx1250) — compile-verified, matrix operand order checked by reading, not run on hardware
//
#include <hip/hip_runtime.h>
#include <hip/hip_bf16.h>

typedef __attribute__((ext_vector_type(16))) __bf16 v16bf;
typedef __attribute__((ext_vector_type(8)))  float  v8f;
typedef __attribute__((ext_vector_type(4)))  unsigned int u32x4;

union BF16Tile { v16bf v; u32x4 q[2]; unsigned int ui[8]; unsigned short us[16]; };
union AccTile  { v8f v; float f[8]; };

// fast f32->bf16 (round-half-up) for hot paths: 2 VALU ops
__device__ __forceinline__ unsigned short f2bf(float f) {
    return (unsigned short)((__builtin_bit_cast(unsigned int, f) + 0x8000u) >> 16);
}
// precise RNE for one-time weight prep
__device__ __forceinline__ unsigned short f2bf_rne(float f) {
    unsigned int u = __builtin_bit_cast(unsigned int, f);
    u += 0x7FFFu + ((u >> 16) & 1u);
    return (unsigned short)(u >> 16);
}
// pack two f32 -> packed bf16 pair {hi[31:16], lo[31:16]} : 2 adds + 1 v_perm_b32
__device__ __forceinline__ unsigned int pack2bf(float lo, float hi) {
    unsigned int ulo = __builtin_bit_cast(unsigned int, lo) + 0x8000u;
    unsigned int uhi = __builtin_bit_cast(unsigned int, hi) + 0x8000u;
    return __builtin_amdgcn_perm(uhi, ulo, 0x07060302u);
}

// 16-bit WMMA fragment K index for element e (0..15) in lane-half hl (0/1)
__device__ __forceinline__ int kpat(int e, int hl) {
    return ((e >> 3) << 4) + (hl << 3) + (e & 7);
}

#define WMMA_BF16(A, B, C) \
    __builtin_amdgcn_wmma_f32_16x16x32_bf16(false, (A), false, (B), (short)0, (C), false, false)

// ---------------------------------------------------------------------------
// Kernel 0: one-time weight prep. Converts Wq/Wk/Wv ([32,256] f32) and
// Wu ([256,32] f32) to bf16 WMMA A-fragment layout:
//   proj:  idx = ((ks*2 + t)*32 + lane)*16 + e  -> W[t*16 + lane%16][ks*32 + kpat(e,lane/16)]
//   Wu:    idx = (tile*32 + lane)*16 + e        -> Wu[tile*16 + lane%16][kpat(e,lane/16)]
// 8192 elements per array; grid 32x256.
// ---------------------------------------------------------------------------
__global__ __launch_bounds__(256) void prep_kernel(
    const float* __restrict__ Wq, const float* __restrict__ Wk,
    const float* __restrict__ Wv, const float* __restrict__ Wu,
    unsigned short* __restrict__ wqf, unsigned short* __restrict__ wkf,
    unsigned short* __restrict__ wvf, unsigned short* __restrict__ wuf)
{
    const int idx  = blockIdx.x * 256 + threadIdx.x;   // 0..8191
    const int e    = idx & 15;
    const int lane = (idx >> 4) & 31;
    const int tile = idx >> 9;                         // 0..15
    const int hl = lane >> 4, lm = lane & 15;
    const int ks = tile >> 1, t = tile & 1;
    const int kk  = ks * 32 + kpat(e, hl);             // 0..255
    const int row = t * 16 + lm;                       // 0..31
    wqf[idx] = f2bf_rne(Wq[row * 256 + kk]);
    wkf[idx] = f2bf_rne(Wk[row * 256 + kk]);
    wvf[idx] = f2bf_rne(Wv[row * 256 + kk]);
    wuf[idx] = f2bf_rne(Wu[(tile * 16 + lm) * 32 + kpat(e, hl)]);
}

// ---------------------------------------------------------------------------
// Kernel 1: fused q/k/v 1x1-conv projections + pixel_unshuffle layout.
// One wave: batch b, patch-position pp (0..15), 16 consecutive patches.
// GEMM [32 x 256] x [256 x 16pix], 8 K-steps, 6 WMMAs/step.
// fq,fk: bf16 [b][512][1600] row-major.  fvT: bf16 [b][1600][512] (transposed).
// ---------------------------------------------------------------------------
__global__ __launch_bounds__(256) void qkv_kernel(
    const float* __restrict__ rgb, const float* __restrict__ depth,
    const unsigned short* __restrict__ wqf, const float* __restrict__ bq,
    const unsigned short* __restrict__ wkf, const float* __restrict__ bk,
    const unsigned short* __restrict__ wvf, const float* __restrict__ bv,
    unsigned short* __restrict__ fq, unsigned short* __restrict__ fk,
    unsigned short* __restrict__ fvT)
{
    const int lane = threadIdx.x & 31;
    const int wave = (blockIdx.x * blockDim.x + threadIdx.x) >> 5;  // 0..12799
    const int hl = lane >> 4, lm = lane & 15;
    const int ntile = wave % 100;
    const int t  = wave / 100;
    const int pp = t & 15;
    const int b  = t >> 4;

    const int n  = ntile * 16 + lm;           // patch index (B-fragment column)
    const int hp = n / 40, wp = n % 40;
    const int h  = hp * 4 + (pp >> 2);
    const int w  = wp * 4 + (pp & 3);
    const long plane = 160L * 160L;
    const float* dep  = depth + (long)b * 256 * plane + (long)h * 160 + w;
    const float* rgbp = rgb   + (long)b * 256 * plane + (long)h * 160 + w;

    AccTile aq0, aq1, ak0, ak1, av0, av1;
    #pragma unroll
    for (int i = 0; i < 8; ++i) {
        aq0.f[i] = 0.f; aq1.f[i] = 0.f; ak0.f[i] = 0.f;
        ak1.f[i] = 0.f; av0.f[i] = 0.f; av1.f[i] = 0.f;
    }

    for (int ks = 0; ks < 8; ++ks) {
        // pixel B-fragments: 16 strided dword loads each, pair-packed to bf16
        BF16Tile Bd, Br;
        #pragma unroll
        for (int p = 0; p < 8; ++p) {
            const int kk = ks * 32 + kpat(2 * p, hl);
            Bd.ui[p] = pack2bf(dep [(long)kk * plane], dep [(long)(kk + 1) * plane]);
            Br.ui[p] = pack2bf(rgbp[(long)kk * plane], rgbp[(long)(kk + 1) * plane]);
        }
        // weight A-fragments: prelayouted, 2 x b128 each
        const long o0 = ((long)(ks * 2    ) * 32 + lane) * 16;
        const long o1 = ((long)(ks * 2 + 1) * 32 + lane) * 16;
        BF16Tile Aq0, Aq1, Ak0, Ak1, Av0, Av1;
        Aq0.q[0] = *reinterpret_cast<const u32x4*>(wqf + o0);
        Aq0.q[1] = *reinterpret_cast<const u32x4*>(wqf + o0 + 8);
        Aq1.q[0] = *reinterpret_cast<const u32x4*>(wqf + o1);
        Aq1.q[1] = *reinterpret_cast<const u32x4*>(wqf + o1 + 8);
        Ak0.q[0] = *reinterpret_cast<const u32x4*>(wkf + o0);
        Ak0.q[1] = *reinterpret_cast<const u32x4*>(wkf + o0 + 8);
        Ak1.q[0] = *reinterpret_cast<const u32x4*>(wkf + o1);
        Ak1.q[1] = *reinterpret_cast<const u32x4*>(wkf + o1 + 8);
        Av0.q[0] = *reinterpret_cast<const u32x4*>(wvf + o0);
        Av0.q[1] = *reinterpret_cast<const u32x4*>(wvf + o0 + 8);
        Av1.q[0] = *reinterpret_cast<const u32x4*>(wvf + o1);
        Av1.q[1] = *reinterpret_cast<const u32x4*>(wvf + o1 + 8);

        aq0.v = WMMA_BF16(Aq0.v, Bd.v, aq0.v);
        aq1.v = WMMA_BF16(Aq1.v, Bd.v, aq1.v);
        ak0.v = WMMA_BF16(Ak0.v, Br.v, ak0.v);
        ak1.v = WMMA_BF16(Ak1.v, Br.v, ak1.v);
        av0.v = WMMA_BF16(Av0.v, Bd.v, av0.v);
        av1.v = WMMA_BF16(Av1.v, Bd.v, av1.v);
    }

    unsigned short* fqb = fq  + (long)b * 512 * 1600;
    unsigned short* fkb = fk  + (long)b * 512 * 1600;
    unsigned short* fvb = fvT + (long)b * 1600 * 512;
    #pragma unroll
    for (int r = 0; r < 8; ++r) {
        const int c0 = r + 8 * hl, c1 = 16 + c0;
        const int d0 = c0 * 16 + pp, d1 = c1 * 16 + pp;   // pixel_unshuffle d index
        fqb[(long)d0 * 1600 + n] = f2bf(aq0.f[r] + bq[c0]);
        fqb[(long)d1 * 1600 + n] = f2bf(aq1.f[r] + bq[c1]);
        fkb[(long)d0 * 1600 + n] = f2bf(ak0.f[r] + bk[c0]);
        fkb[(long)d1 * 1600 + n] = f2bf(ak1.f[r] + bk[c1]);
        fvb[(long)n * 512 + d0]  = f2bf(av0.f[r] + bv[c0]);
        fvb[(long)n * 512 + d1]  = f2bf(av1.f[r] + bv[c1]);
    }
}

// ---------------------------------------------------------------------------
// Kernel 2: scores[b] = fq[b] (512x1600) x fk[b]^T / sqrt(512), f32 out.
// One wave computes a 16x64 strip (A reused across 4 B tiles), K = 50 steps.
// ---------------------------------------------------------------------------
__global__ __launch_bounds__(256) void scores_kernel(
    const unsigned short* __restrict__ fq, const unsigned short* __restrict__ fk,
    float* __restrict__ scores)
{
    const int lane = threadIdx.x & 31;
    const int wave = (blockIdx.x * blockDim.x + threadIdx.x) >> 5;  // 0..2047
    const int hl = lane >> 4, lm = lane & 15;
    const int cs = wave & 7;
    const int t  = wave >> 3;
    const int mt = t & 31;
    const int b  = t >> 5;

    const unsigned short* Arow = fq + ((long)b * 512 + mt * 16 + lm) * 1600;
    AccTile acc[4];
    #pragma unroll
    for (int s = 0; s < 4; ++s)
        #pragma unroll
        for (int r = 0; r < 8; ++r) acc[s].f[r] = 0.f;

    for (int ks = 0; ks < 50; ++ks) {
        const int kb = ks * 32;
        BF16Tile A;
        A.q[0] = *reinterpret_cast<const u32x4*>(Arow + kb + hl * 8);
        A.q[1] = *reinterpret_cast<const u32x4*>(Arow + kb + 16 + hl * 8);
        #pragma unroll
        for (int s = 0; s < 4; ++s) {
            const unsigned short* Brow = fk + ((long)b * 512 + cs * 64 + s * 16 + lm) * 1600;
            BF16Tile Bt;
            Bt.q[0] = *reinterpret_cast<const u32x4*>(Brow + kb + hl * 8);
            Bt.q[1] = *reinterpret_cast<const u32x4*>(Brow + kb + 16 + hl * 8);
            acc[s].v = WMMA_BF16(A.v, Bt.v, acc[s].v);
        }
    }

    float* sb = scores + (long)b * 512 * 512;
    const float scl = 0.0441941738241592f;   // 1/sqrt(512)
    #pragma unroll
    for (int s = 0; s < 4; ++s)
        #pragma unroll
        for (int r = 0; r < 8; ++r) {
            const int row = mt * 16 + r + 8 * hl;
            const int col = cs * 64 + s * 16 + lm;
            sb[(long)row * 512 + col] = acc[s].f[r] * scl;
        }
}

// ---------------------------------------------------------------------------
// Kernel 3: row softmax over 512 cols; one wave per row; bf16 attn out.
// ---------------------------------------------------------------------------
__global__ __launch_bounds__(256) void softmax_kernel(
    const float* __restrict__ scores, unsigned short* __restrict__ attn)
{
    const int lane = threadIdx.x & 31;
    const int row  = (blockIdx.x * blockDim.x + threadIdx.x) >> 5;  // 0..4095
    const float* src = scores + (long)row * 512;
    float v[16];
    float m = -3.0e38f;
    #pragma unroll
    for (int i = 0; i < 16; ++i) { v[i] = src[i * 32 + lane]; m = fmaxf(m, v[i]); }
    #pragma unroll
    for (int off = 16; off > 0; off >>= 1) m = fmaxf(m, __shfl_xor(m, off, 32));
    float s = 0.f;
    #pragma unroll
    for (int i = 0; i < 16; ++i) { v[i] = __expf(v[i] - m); s += v[i]; }
    #pragma unroll
    for (int off = 16; off > 0; off >>= 1) s += __shfl_xor(s, off, 32);
    const float inv = 1.0f / s;
    unsigned short* dst = attn + (long)row * 512;
    #pragma unroll
    for (int i = 0; i < 16; ++i) dst[i * 32 + lane] = f2bf(v[i] * inv);
}

// ---------------------------------------------------------------------------
// Kernel 4: out_mid[b] = attn[b] (512x512 bf16) x fv[b] (512x1600), K = 16 steps.
// Stored permuted: om[b][n][pp*32 + c] with d = c*16 + pp, for kernel 5's B frag.
// ---------------------------------------------------------------------------
__global__ __launch_bounds__(256) void av_kernel(
    const unsigned short* __restrict__ attn, const unsigned short* __restrict__ fvT,
    unsigned short* __restrict__ om)
{
    const int lane = threadIdx.x & 31;
    const int wave = (blockIdx.x * blockDim.x + threadIdx.x) >> 5;  // 0..6399
    const int hl = lane >> 4, lm = lane & 15;
    const int ns = wave % 25;
    const int t  = wave / 25;
    const int mt = t & 31;
    const int b  = t >> 5;

    const unsigned short* Arow = attn + ((long)b * 512 + mt * 16 + lm) * 512;
    AccTile acc[4];
    #pragma unroll
    for (int s = 0; s < 4; ++s)
        #pragma unroll
        for (int r = 0; r < 8; ++r) acc[s].f[r] = 0.f;

    for (int ks = 0; ks < 16; ++ks) {
        const int kb = ks * 32;
        BF16Tile A;
        A.q[0] = *reinterpret_cast<const u32x4*>(Arow + kb + hl * 8);
        A.q[1] = *reinterpret_cast<const u32x4*>(Arow + kb + 16 + hl * 8);
        #pragma unroll
        for (int s = 0; s < 4; ++s) {
            const unsigned short* Brow = fvT + ((long)b * 1600 + ns * 64 + s * 16 + lm) * 512;
            BF16Tile Bt;
            Bt.q[0] = *reinterpret_cast<const u32x4*>(Brow + kb + hl * 8);
            Bt.q[1] = *reinterpret_cast<const u32x4*>(Brow + kb + 16 + hl * 8);
            acc[s].v = WMMA_BF16(A.v, Bt.v, acc[s].v);
        }
    }

    #pragma unroll
    for (int s = 0; s < 4; ++s)
        #pragma unroll
        for (int r = 0; r < 8; ++r) {
            const int d = mt * 16 + r + 8 * hl;
            const int n = ns * 64 + s * 16 + lm;
            om[((long)b * 1600 + n) * 512 + (d & 15) * 32 + (d >> 4)] = f2bf(acc[s].f[r]);
        }
}

// ---------------------------------------------------------------------------
// Kernel 5: pixel_shuffle + output 1x1 conv: Wu[256,32] x 32-vec per pixel.
// One wave: 64 output channels x 16 consecutive pixels, single K=32 WMMA/tile.
// A-fragments preconverted by prep_kernel (2 x b128 per tile).
// ---------------------------------------------------------------------------
__global__ __launch_bounds__(256) void outproj_kernel(
    const unsigned short* __restrict__ om, const unsigned short* __restrict__ wuf,
    const float* __restrict__ bu, float* __restrict__ out)
{
    const int lane = threadIdx.x & 31;
    const int wave = (blockIdx.x * blockDim.x + threadIdx.x) >> 5;  // 0..51199
    const int hl = lane >> 4, lm = lane & 15;
    const int ms  = wave & 3;
    const int t   = wave >> 2;
    const int grp = t % 1600;
    const int b   = t / 1600;
    const int h  = grp / 10;
    const int w0 = (grp % 10) * 16;
    const int w  = w0 + lm;
    const int pp = (h & 3) * 4 + (w & 3);
    const int n  = (h >> 2) * 40 + (w >> 2);

    BF16Tile Bt;
    const unsigned short* Brow = om + ((long)b * 1600 + n) * 512 + pp * 32;
    Bt.q[0] = *reinterpret_cast<const u32x4*>(Brow + hl * 8);
    Bt.q[1] = *reinterpret_cast<const u32x4*>(Brow + 16 + hl * 8);

    AccTile acc[4];
    #pragma unroll
    for (int s = 0; s < 4; ++s) {
        #pragma unroll
        for (int r = 0; r < 8; ++r) acc[s].f[r] = 0.f;
        BF16Tile A;
        const unsigned short* Ap = wuf + ((long)(ms * 4 + s) * 32 + lane) * 16;
        A.q[0] = *reinterpret_cast<const u32x4*>(Ap);
        A.q[1] = *reinterpret_cast<const u32x4*>(Ap + 8);
        acc[s].v = WMMA_BF16(A.v, Bt.v, acc[s].v);
    }

    #pragma unroll
    for (int s = 0; s < 4; ++s)
        #pragma unroll
        for (int r = 0; r < 8; ++r) {
            const int co = ms * 64 + s * 16 + r + 8 * hl;
            out[((long)b * 256 + co) * 25600 + (long)h * 160 + w0 + lm] =
                acc[s].f[r] + bu[co];
        }
}

extern "C" void kernel_launch(void* const* d_in, const int* in_sizes, int n_in,
                              void* d_out, int out_size, void* d_ws, size_t ws_size,
                              hipStream_t stream) {
    const float* rgb   = (const float*)d_in[0];
    const float* depth = (const float*)d_in[1];
    const float* Wq = (const float*)d_in[2];
    const float* bq = (const float*)d_in[3];
    const float* Wk = (const float*)d_in[4];
    const float* bk = (const float*)d_in[5];
    const float* Wv = (const float*)d_in[6];
    const float* bv = (const float*)d_in[7];
    const float* Wu = (const float*)d_in[8];
    const float* bu = (const float*)d_in[9];
    float* out = (float*)d_out;

    // workspace carve-up (all 16B aligned)
    unsigned short* fq  = (unsigned short*)d_ws;
    unsigned short* fk  = fq  + (size_t)8 * 512 * 1600;
    unsigned short* fvT = fk  + (size_t)8 * 512 * 1600;
    float* scores       = (float*)(fvT + (size_t)8 * 1600 * 512);
    unsigned short* attn = (unsigned short*)(scores + (size_t)8 * 512 * 512);
    unsigned short* om   = attn + (size_t)8 * 512 * 512;
    unsigned short* wqf  = om  + (size_t)8 * 1600 * 512;
    unsigned short* wkf  = wqf + 8192;
    unsigned short* wvf  = wkf + 8192;
    unsigned short* wuf  = wvf + 8192;

    prep_kernel   <<<  32, 256, 0, stream>>>(Wq, Wk, Wv, Wu, wqf, wkf, wvf, wuf);
    qkv_kernel    <<<1600, 256, 0, stream>>>(rgb, depth, wqf, bq, wkf, bk, wvf, bv, fq, fk, fvT);
    scores_kernel <<< 256, 256, 0, stream>>>(fq, fk, scores);
    softmax_kernel<<< 512, 256, 0, stream>>>(scores, attn);
    av_kernel     <<< 800, 256, 0, stream>>>(attn, fvT, om);
    outproj_kernel<<<6400, 256, 0, stream>>>(om, wuf, bu, out);
}
